// InferenceModel_34033320853994
// MI455X (gfx1250) — matrix operand order repaired and checked by reading, not compiled, not run on hardware
//
#include <hip/hip_runtime.h>

#define BB   1024
#define TT   2048
#define NT   64                       // batch tiles of 16
#define BT10 ((size_t)BB * (size_t)TT * 10)

typedef float    v8f  __attribute__((ext_vector_type(8)));
typedef _Float16 v16h __attribute__((ext_vector_type(16)));
typedef _Float16 v8h  __attribute__((ext_vector_type(8)));

// D = A(16x32 f16) x B(32x16 f16) + C(16x16 f32); K=16..31 zero-padded.
__device__ __forceinline__ v8f wmma16(v16h a, v16h b, v8f c) {
  return __builtin_amdgcn_wmma_f32_16x16x32_f16(false, a, false, b, (short)0, c,
                                                false, false);
}

__device__ __forceinline__ float fsig(float x) {
  return __builtin_amdgcn_rcpf(1.0f + __expf(-x));
}
__device__ __forceinline__ float ftanh(float x) {
  x = fminf(fmaxf(x, -15.0f), 15.0f);
  const float e = __expf(2.0f * x);
  return (e - 1.0f) * __builtin_amdgcn_rcpf(e + 1.0f);
}

// ---------------------------------------------------------------------------
// Pass 1: forward & backward GRU over R (RET=1).  One wave32 per
// (direction, 16-batch tile).  Hidden matmul = 3 x v_wmma_f32_16x16x32_f16.
// Emits h_t in B-operand layout (f16, [b][k], 512B/step) to the workspace.
// ---------------------------------------------------------------------------
__global__ __launch_bounds__(32)
void gru_fb_pass(const float* __restrict__ R,
                 const float* __restrict__ wih_f, const float* __restrict__ whh_f,
                 const float* __restrict__ bih_f, const float* __restrict__ bhh_f,
                 const float* __restrict__ wih_b, const float* __restrict__ whh_b,
                 const float* __restrict__ bih_b, const float* __restrict__ bhh_b,
                 _Float16* __restrict__ hB) {
  const int lane = threadIdx.x & 31;
  const int dir  = blockIdx.x >> 6;   // 0 = forward, 1 = backward
  const int tile = blockIdx.x & 63;
  const int m    = lane & 15;
  const int off8 = (lane < 16) ? 0 : 8;

  const float* wih = dir ? wih_b : wih_f;
  const float* whh = dir ? whh_b : whh_f;
  const float* bih = dir ? bih_b : bih_f;
  const float* bhh = dir ? bhh_b : bhh_f;

  __shared__ __align__(32) _Float16 sm[16 * 16 + 16];
  _Float16* hrow = sm;          // [b][k] f16 tile for B-layout rebuild
  _Float16* zrow = sm + 256;    // 32B of zeros for lanes>=16 (K=16..31 pad)
  if (lane < 16) zrow[lane] = (_Float16)0.0f;
  __syncthreads();

  // Weight-stationary A tiles (whh, 3 gate chunks r/z/n) + bias C tiles.
  v16h Awhh[3];
  v8f  Cbhh[3];
  float wv[3][8], bv[3][8];
  #pragma unroll
  for (int gc = 0; gc < 3; ++gc) {
    const int row = gc * 16 + m;
    v16h a;
    #pragma unroll
    for (int e = 0; e < 16; ++e)
      a[e] = (lane < 16) ? (_Float16)whh[row * 16 + e] : (_Float16)0.0f;
    Awhh[gc] = a;
    v8f c;
    #pragma unroll
    for (int v = 0; v < 8; ++v) {
      const int g = gc * 16 + off8 + v;
      c[v]      = bhh[g];
      wv[gc][v] = wih[g];      // RET == 1 -> input proj is an outer product
      bv[gc][v] = bih[g];
    }
    Cbhh[gc] = c;
  }

  const float* Rrow  = R + (size_t)(tile * 16 + m) * TT;   // R is [B][T]
  _Float16*    hOut  = hB + (size_t)(dir * NT + tile) * TT * 256;

  v16h  Bh = {};                 // h0 = 0 (B layout)
  float hD[8] = {0, 0, 0, 0, 0, 0, 0, 0};   // h (D layout: M=hid, N=batch)

  for (int t = 0; t < TT; ++t) {
    const int ts = dir ? (TT - 1 - t) : t;
    int tp = dir ? (ts - 64) : (ts + 64);
    tp = tp < 0 ? 0 : (tp > TT - 1 ? TT - 1 : tp);
    __builtin_prefetch(Rrow + tp, 0, 3);     // global_prefetch_b8
    const float Rb = Rrow[ts];

    // hg = whh @ h + bhh   (bias rides in as the C operand)
    v8f g0 = wmma16(Awhh[0], Bh, Cbhh[0]);
    v8f g1 = wmma16(Awhh[1], Bh, Cbhh[1]);
    v8f g2 = wmma16(Awhh[2], Bh, Cbhh[2]);

    v8h p;
    #pragma unroll
    for (int v = 0; v < 8; ++v) {
      const float xr = fmaf(wv[0][v], Rb, bv[0][v]);
      const float xz = fmaf(wv[1][v], Rb, bv[1][v]);
      const float xn = fmaf(wv[2][v], Rb, bv[2][v]);
      const float r  = fsig(xr + g0[v]);
      const float u  = fsig(xz + g1[v]);
      const float n  = ftanh(xn + r * g2[v]);
      hD[v] = (1.0f - u) * n + u * hD[v];
      p[v]  = (_Float16)hD[v];
    }

    // D-layout -> B-layout bounce through LDS; also publish B-layout to HBM.
    *(v8h*)(hrow + m * 16 + off8) = p;
    *(v8h*)(hOut + (size_t)ts * 256 + m * 16 + off8) = p;
    __syncthreads();
    Bh = *(const v16h*)((lane < 16) ? (hrow + lane * 16) : zrow);
    __syncthreads();
  }
}

// ---------------------------------------------------------------------------
// Pass 2: sequential latent scan.  One wave32 per 16-batch tile.
// Per step: 18 x v_wmma_f32_16x16x32_f16 (xg chain, hg, 2 MLPs interleaved).
// ---------------------------------------------------------------------------
__global__ __launch_bounds__(32)
void z_scan_pass(const float* __restrict__ eps,
                 const float* __restrict__ wih_z, const float* __restrict__ whh_z,
                 const float* __restrict__ bih_z, const float* __restrict__ bhh_z,
                 const float* __restrict__ mw1, const float* __restrict__ mb1,
                 const float* __restrict__ mw2, const float* __restrict__ mb2,
                 const float* __restrict__ mwo, const float* __restrict__ mbo,
                 const float* __restrict__ sw1, const float* __restrict__ sb1,
                 const float* __restrict__ sw2, const float* __restrict__ sb2,
                 const float* __restrict__ swo, const float* __restrict__ sbo,
                 const _Float16* __restrict__ hB,
                 float* __restrict__ out) {
  const int lane = threadIdx.x & 31;
  const int tile = blockIdx.x;
  const int m    = lane & 15;
  const int off8 = (lane < 16) ? 0 : 8;

  __shared__ __align__(32) _Float16 sm[16 * 16 * 6 + 16];
  _Float16* sH   = sm;
  _Float16* s1m  = sm + 256;
  _Float16* s1s  = sm + 512;
  _Float16* s2m  = sm + 768;
  _Float16* s2s  = sm + 1024;
  _Float16* sZ   = sm + 1280;
  _Float16* zrow = sm + 1536;
  if (lane < 16) zrow[lane] = (_Float16)0.0f;
  __syncthreads();

  // --- A-layout weights + C-operand biases ---------------------------------
  v16h Awz[3], Awf[3], Awb[3], Ahh[3];
  v8f  Cbi[3], Cbh[3];
  #pragma unroll
  for (int gc = 0; gc < 3; ++gc) {
    const int row = gc * 16 + m;
    v16h az, af, ab, ah;
    #pragma unroll
    for (int e = 0; e < 16; ++e) {
      if (lane < 16) {
        const int k = e;
        az[e] = (k < 10) ? (_Float16)wih_z[row * 42 + k] : (_Float16)0.0f;
        af[e] = (_Float16)wih_z[row * 42 + 10 + k];
        ab[e] = (_Float16)wih_z[row * 42 + 26 + k];
        ah[e] = (_Float16)whh_z[row * 16 + k];
      } else {
        az[e] = af[e] = ab[e] = ah[e] = (_Float16)0.0f;
      }
    }
    Awz[gc] = az; Awf[gc] = af; Awb[gc] = ab; Ahh[gc] = ah;
    v8f ci, ch;
    #pragma unroll
    for (int v = 0; v < 8; ++v) {
      const int g = gc * 16 + off8 + v;
      ci[v] = bih_z[g];
      ch[v] = bhh_z[g];
    }
    Cbi[gc] = ci; Cbh[gc] = ch;
  }
  v16h A1m, A2m, Aom, A1s, A2s, Aos;
  v8f  C1m, C2m, Com, C1s, C2s, Cos;
  #pragma unroll
  for (int e = 0; e < 16; ++e) {
    if (lane < 16) {
      const int k = e;
      A1m[e] = (_Float16)mw1[m * 16 + k];
      A2m[e] = (_Float16)mw2[m * 16 + k];
      A1s[e] = (_Float16)sw1[m * 16 + k];
      A2s[e] = (_Float16)sw2[m * 16 + k];
      Aom[e] = (m < 10) ? (_Float16)mwo[m * 16 + k] : (_Float16)0.0f;
      Aos[e] = (m < 10) ? (_Float16)swo[m * 16 + k] : (_Float16)0.0f;
    } else {
      A1m[e] = A2m[e] = A1s[e] = A2s[e] = Aom[e] = Aos[e] = (_Float16)0.0f;
    }
  }
  #pragma unroll
  for (int v = 0; v < 8; ++v) {
    const int o = off8 + v;
    C1m[v] = mb1[o]; C2m[v] = mb2[o];
    C1s[v] = sb1[o]; C2s[v] = sb2[o];
    Com[v] = (o < 10) ? mbo[o] : 0.0f;
    Cos[v] = (o < 10) ? sbo[o] : 0.0f;
  }

  const _Float16* hfP = hB + (size_t)tile * TT * 256;
  const _Float16* hbP = hB + (size_t)(NT + tile) * TT * 256;
  const int       b   = tile * 16 + m;
  const float*    epsP = eps + (size_t)b * TT * 10;
  float* zO = out + (size_t)b * TT * 10;
  float* mO = zO + BT10;
  float* sO = zO + 2 * BT10;

  v16h  Bz = {}, Bh = {};
  float hD[8] = {0, 0, 0, 0, 0, 0, 0, 0};

  for (int t = 0; t < TT; ++t) {
    {
      const int tp = (t + 8 > TT - 1) ? (TT - 1) : (t + 8);
      __builtin_prefetch(hfP + (size_t)tp * 256 + lane * 16, 0, 3);
      __builtin_prefetch(hbP + (size_t)tp * 256 + lane * 16, 0, 3);
      __builtin_prefetch(epsP + (size_t)tp * 10, 0, 3);
    }
    v16h Bhf = {}, Bhb = {};
    if (lane < 16) {
      Bhf = *(const v16h*)(hfP + (size_t)t * 256 + lane * 16);
      Bhb = *(const v16h*)(hbP + (size_t)t * 256 + lane * 16);
    }

    // xg = Wf@hf + Wb@hb + Wz@z + bih_z  (chained through the C accumulator)
    v8f x0 = wmma16(Awf[0], Bhf, Cbi[0]);
    v8f x1 = wmma16(Awf[1], Bhf, Cbi[1]);
    v8f x2 = wmma16(Awf[2], Bhf, Cbi[2]);
    x0 = wmma16(Awb[0], Bhb, x0);
    x1 = wmma16(Awb[1], Bhb, x1);
    x2 = wmma16(Awb[2], Bhb, x2);
    x0 = wmma16(Awz[0], Bz, x0);
    x1 = wmma16(Awz[1], Bz, x1);
    x2 = wmma16(Awz[2], Bz, x2);
    // hg = whh_z @ h + bhh_z
    v8f h0 = wmma16(Ahh[0], Bh, Cbh[0]);
    v8f h1 = wmma16(Ahh[1], Bh, Cbh[1]);
    v8f h2 = wmma16(Ahh[2], Bh, Cbh[2]);

    v8h p;
    #pragma unroll
    for (int v = 0; v < 8; ++v) {
      const float r = fsig(x0[v] + h0[v]);
      const float u = fsig(x1[v] + h1[v]);
      const float n = ftanh(x2[v] + r * h2[v]);
      hD[v] = (1.0f - u) * n + u * hD[v];
      p[v]  = (_Float16)hD[v];
    }
    *(v8h*)(sH + m * 16 + off8) = p;
    __syncthreads();
    Bh = *(const v16h*)((lane < 16) ? (sH + lane * 16) : zrow);
    __syncthreads();

    // --- MLP layer 1 (mu & sigma interleaved for ILP) ---
    v8f d1m = wmma16(A1m, Bh, C1m);
    v8f d1s = wmma16(A1s, Bh, C1s);
    v8h pm, ps;
    #pragma unroll
    for (int v = 0; v < 8; ++v) {
      pm[v] = (_Float16)fmaxf(d1m[v], 0.0f);
      ps[v] = (_Float16)fmaxf(d1s[v], 0.0f);
    }
    *(v8h*)(s1m + m * 16 + off8) = pm;
    *(v8h*)(s1s + m * 16 + off8) = ps;
    __syncthreads();
    v16h B1m = *(const v16h*)((lane < 16) ? (s1m + lane * 16) : zrow);
    v16h B1s = *(const v16h*)((lane < 16) ? (s1s + lane * 16) : zrow);
    __syncthreads();

    // --- MLP layer 2 ---
    v8f d2m = wmma16(A2m, B1m, C2m);
    v8f d2s = wmma16(A2s, B1s, C2s);
    #pragma unroll
    for (int v = 0; v < 8; ++v) {
      pm[v] = (_Float16)fmaxf(d2m[v], 0.0f);
      ps[v] = (_Float16)fmaxf(d2s[v], 0.0f);
    }
    *(v8h*)(s2m + m * 16 + off8) = pm;
    *(v8h*)(s2s + m * 16 + off8) = ps;
    __syncthreads();
    v16h B2m = *(const v16h*)((lane < 16) ? (s2m + lane * 16) : zrow);
    v16h B2s = *(const v16h*)((lane < 16) ? (s2s + lane * 16) : zrow);
    __syncthreads();

    // --- output heads + rsample ---
    v8f muD = wmma16(Aom, B2m, Com);
    v8f lsD = wmma16(Aos, B2s, Cos);

    const float* ept = epsP + (size_t)t * 10;
    v8h pz;
    #pragma unroll
    for (int v = 0; v < 8; ++v) {
      const int   l  = off8 + v;
      const float e  = ept[(l < 10) ? l : 9];
      const float mu = muD[v];
      const float sg = __expf(lsD[v]);
      const float z  = fmaf(sg, e, mu);
      if (l < 10) {
        zO[(size_t)t * 10 + l] = z;
        mO[(size_t)t * 10 + l] = mu;
        sO[(size_t)t * 10 + l] = sg;
      }
      pz[v] = (l < 10) ? (_Float16)z : (_Float16)0.0f;  // zero K=10..15 pad
    }
    *(v8h*)(sZ + m * 16 + off8) = pz;
    __syncthreads();
    Bz = *(const v16h*)((lane < 16) ? (sZ + lane * 16) : zrow);
    __syncthreads();
  }
}

// ---------------------------------------------------------------------------
extern "C" void kernel_launch(void* const* d_in, const int* in_sizes, int n_in,
                              void* d_out, int out_size, void* d_ws, size_t ws_size,
                              hipStream_t stream) {
  (void)in_sizes; (void)n_in; (void)out_size; (void)ws_size;
  const float* R     = (const float*)d_in[0];
  const float* eps   = (const float*)d_in[1];
  const float* wih_f = (const float*)d_in[2];
  const float* whh_f = (const float*)d_in[3];
  const float* bih_f = (const float*)d_in[4];
  const float* bhh_f = (const float*)d_in[5];
  const float* wih_b = (const float*)d_in[6];
  const float* whh_b = (const float*)d_in[7];
  const float* bih_b = (const float*)d_in[8];
  const float* bhh_b = (const float*)d_in[9];
  const float* wih_z = (const float*)d_in[10];
  const float* whh_z = (const float*)d_in[11];
  const float* bih_z = (const float*)d_in[12];
  const float* bhh_z = (const float*)d_in[13];
  const float* mw1 = (const float*)d_in[14];
  const float* mb1 = (const float*)d_in[15];
  const float* mw2 = (const float*)d_in[16];
  const float* mb2 = (const float*)d_in[17];
  const float* mwo = (const float*)d_in[18];
  const float* mbo = (const float*)d_in[19];
  const float* sw1 = (const float*)d_in[20];
  const float* sb1 = (const float*)d_in[21];
  const float* sw2 = (const float*)d_in[22];
  const float* sb2 = (const float*)d_in[23];
  const float* swo = (const float*)d_in[24];
  const float* sbo = (const float*)d_in[25];

  // Workspace: B-layout f16 h-state, [2 dirs][64 tiles][2048 steps][512B] = 128 MiB.
  _Float16* hB = (_Float16*)d_ws;

  gru_fb_pass<<<dim3(128), dim3(32), 0, stream>>>(
      R, wih_f, whh_f, bih_f, bhh_f, wih_b, whh_b, bih_b, bhh_b, hB);
  z_scan_pass<<<dim3(64), dim3(32), 0, stream>>>(
      eps, wih_z, whh_z, bih_z, bhh_z,
      mw1, mb1, mw2, mb2, mwo, mbo,
      sw1, sb1, sw2, sb2, swo, sbo,
      hB, (float*)d_out);
}
